// LiquidSpikingNetwork_64098091926008
// MI455X (gfx1250) — matrix operand order repaired and checked by reading, not compiled, not run on hardware
//
#include <hip/hip_runtime.h>

// CDNA5 / gfx1250, wave32. Fully fused LiquidSpikingNetwork forward.
// One workgroup (8 waves, 256 threads) = one 16-token tile; each wave owns a
// narrow column slice so persistent SNN state stays ~120 VGPRs/wave.
// All GEMMs via v_wmma_f32_16x16x32_f16 (f16 inputs, f32 accumulate).
// __launch_bounds__(256, 4): >=4 waves/SIMD -> <=256-VGPR budget, no spills/MSB.

typedef __attribute__((ext_vector_type(16))) _Float16 v16h;
typedef __attribute__((ext_vector_type(8)))  _Float16 v8h;
typedef __attribute__((ext_vector_type(8)))  float    v8f;

#define WMMA(a, b, c) \
  __builtin_amdgcn_wmma_f32_16x16x32_f16(false, (a), false, (b), (short)0, (c), false, false)

#define LDSTR 536  // f16 row stride: 1072 B = 16-B aligned, 12-bank row rotation (conflict-free)

// A fragment (16x32 f16) from row-major f32 source. Per lane: K chunks {base, base+8}.
__device__ __forceinline__ v16h a_frag_f32(const float* __restrict__ p) {
  v8f f0 = *(const v8f*)p;
  v8f f1 = *(const v8f*)(p + 8);
  v8h h0 = __builtin_convertvector(f0, v8h);
  v8h h1 = __builtin_convertvector(f1, v8h);
  return __builtin_shufflevector(h0, h1, 0,1,2,3,4,5,6,7,8,9,10,11,12,13,14,15);
}

// A fragment (16x32 f16) from row-major f16 LDS. Per lane: K chunks {base, base+8}.
__device__ __forceinline__ v16h a_frag_lds(const _Float16* p) {
  v8h h0 = *(const v8h*)p;
  v8h h1 = *(const v8h*)(p + 8);
  return __builtin_shufflevector(h0, h1, 0,1,2,3,4,5,6,7,8,9,10,11,12,13,14,15);
}

__global__ __launch_bounds__(256, 4)
void lsn_fused(const float* __restrict__ X,
               const float* __restrict__ b1,  const float* __restrict__ b2,
               const float* __restrict__ bfv, const float* __restrict__ bfu,
               const float* __restrict__ lng, const float* __restrict__ lnb,
               const _Float16* __restrict__ wh, float* __restrict__ out)
{
  const _Float16* W1h  = wh;                                    // [512,512]
  const _Float16* W2h  = wh + 512*512;                          // [256,512]
  const _Float16* Wfh  = wh + 512*512 + 256*512;                // [256,256]
  const _Float16* Wfuh = wh + 512*512 + 256*512 + 256*256;      // [512,512]

  __shared__ _Float16 lsh[16 * LDSTR];   // spk1 [16,512] -> later combined [16,512]
  __shared__ float redS[16];
  __shared__ float redQ[16];

  const int tid  = threadIdx.x;
  const int wave = tid >> 5;     // 0..7
  const int lane = tid & 31;
  const int l16  = lane & 15;
  const int hi   = lane >> 4;    // 0/1: selects K-halves (A/B) and row-halves (C)
  const int rH   = hi << 3;      // C-layout row offset: 0 or 8
  const int rowBase = blockIdx.x * 16;

  if (tid < 16) { redS[tid] = 0.0f; redQ[tid] = 0.0f; }

  // ---------------- Phase 1: mem1 = X @ W1^T + b1  (wave owns H1 cols [64w,64w+64))
  float b1c[4];
#pragma unroll
  for (int t = 0; t < 4; ++t) b1c[t] = b1[wave*64 + t*16 + l16];

  v8f m1[4];
#pragma unroll
  for (int t = 0; t < 4; ++t)
#pragma unroll
    for (int r = 0; r < 8; ++r) m1[t][r] = b1c[t];

  for (int k0 = 0; k0 < 512; k0 += 32) {
    v16h a = a_frag_f32(X + (size_t)(rowBase + l16) * 512 + k0 + hi*16);
#pragma unroll
    for (int t = 0; t < 4; ++t) {
      const _Float16* bp = W1h + (size_t)(wave*64 + t*16 + l16) * 512 + k0 + hi*16;
      m1[t] = WMMA(a, *(const v16h*)bp, m1[t]);
    }
  }

  // ---------------- Phase 2: SNN recurrence (wave owns SU cols [32w,32w+32))
  float b2c[2];
#pragma unroll
  for (int t = 0; t < 2; ++t) b2c[t] = b2[wave*32 + t*16 + l16];

  v8f m2[2], acc[2];
#pragma unroll
  for (int t = 0; t < 2; ++t)
#pragma unroll
    for (int r = 0; r < 8; ++r) { m2[t][r] = 0.0f; acc[t][r] = 0.0f; }

  unsigned s1m = 0u;  // spk1 bits (32 elems/thread)
  unsigned s2m = 0u;  // spk2 bits (16 elems/thread)

  for (int st = 0; st < 16; ++st) {
    // Leaky update of mem1 (t>0: cur1 = b1; t==0: mem1 already = cur0), spike, share via LDS
    unsigned n1m = 0u;
#pragma unroll
    for (int t = 0; t < 4; ++t) {
#pragma unroll
      for (int r = 0; r < 8; ++r) {
        const int e = t*8 + r;
        float m = m1[t][r];
        const float upd = 0.95f*m + b1c[t] - (((s1m >> e) & 1u) ? 1.0f : 0.0f);
        m = (st > 0) ? upd : m;
        const bool sp = (m - 1.0f) > 0.0f;
        m1[t][r] = m;
        if (sp) n1m |= (1u << e);
        lsh[(r + rH) * LDSTR + wave*64 + t*16 + l16] = sp ? (_Float16)1.0f : (_Float16)0.0f;
      }
    }
    s1m = n1m;
    __syncthreads();

    // cur2 = spk1 @ W2^T + b2
    v8f c2[2];
#pragma unroll
    for (int t = 0; t < 2; ++t)
#pragma unroll
      for (int r = 0; r < 8; ++r) c2[t][r] = b2c[t];

    for (int k0 = 0; k0 < 512; k0 += 32) {
      v16h a = a_frag_lds(&lsh[l16 * LDSTR + k0 + hi*16]);
#pragma unroll
      for (int t = 0; t < 2; ++t) {
        const _Float16* bp = W2h + (size_t)(wave*32 + t*16 + l16) * 512 + k0 + hi*16;
        c2[t] = WMMA(a, *(const v16h*)bp, c2[t]);
      }
    }

    // mem2 leaky update + spike + accumulate spike counts
    unsigned n2m = 0u;
#pragma unroll
    for (int t = 0; t < 2; ++t) {
#pragma unroll
      for (int r = 0; r < 8; ++r) {
        const int e = t*8 + r;
        float m = 0.95f*m2[t][r] + c2[t][r] - (((s2m >> e) & 1u) ? 1.0f : 0.0f);
        const bool sp = (m - 1.0f) > 0.0f;
        m2[t][r] = m;
        if (sp) { n2m |= (1u << e); acc[t][r] += 1.0f; }
      }
    }
    s2m = n2m;
    __syncthreads();  // protect next step's spk1 overwrite (and Phase-3 reuse)
  }

  // ---------------- Phase 3a: spike_features = mean -> LDS cols [256,512)
#pragma unroll
  for (int t = 0; t < 2; ++t)
#pragma unroll
    for (int r = 0; r < 8; ++r)
      lsh[(r + rH) * LDSTR + 256 + wave*32 + t*16 + l16] = (_Float16)(acc[t][r] * 0.0625f);
  __syncthreads();

  // ---------------- Phase 3b: liquid_out = sf @ Wf^T + bf -> LDS cols [0,256)
  float bfc[2];
#pragma unroll
  for (int t = 0; t < 2; ++t) bfc[t] = bfv[wave*32 + t*16 + l16];
  v8f lo[2];
#pragma unroll
  for (int t = 0; t < 2; ++t)
#pragma unroll
    for (int r = 0; r < 8; ++r) lo[t][r] = bfc[t];

  for (int k0 = 0; k0 < 256; k0 += 32) {
    v16h a = a_frag_lds(&lsh[l16 * LDSTR + 256 + k0 + hi*16]);
#pragma unroll
    for (int t = 0; t < 2; ++t) {
      const _Float16* bp = Wfh + (size_t)(wave*32 + t*16 + l16) * 256 + k0 + hi*16;
      lo[t] = WMMA(a, *(const v16h*)bp, lo[t]);
    }
  }
#pragma unroll
  for (int t = 0; t < 2; ++t)   // disjoint from cols [256,512) still being read -> no race
#pragma unroll
    for (int r = 0; r < 8; ++r)
      lsh[(r + rH) * LDSTR + wave*32 + t*16 + l16] = (_Float16)lo[t][r];
  __syncthreads();

  // ---------------- Phase 4: y = combined @ Wfu^T + bfu (wave owns F cols [64w,64w+64))
  float bfuc[4], gc[4], bc[4];
#pragma unroll
  for (int t = 0; t < 4; ++t) {
    const int col = wave*64 + t*16 + l16;
    bfuc[t] = bfu[col]; gc[t] = lng[col]; bc[t] = lnb[col];
  }
  v8f y[4];
#pragma unroll
  for (int t = 0; t < 4; ++t)
#pragma unroll
    for (int r = 0; r < 8; ++r) y[t][r] = bfuc[t];

  for (int k0 = 0; k0 < 512; k0 += 32) {
    v16h a = a_frag_lds(&lsh[l16 * LDSTR + k0 + hi*16]);
#pragma unroll
    for (int t = 0; t < 4; ++t) {
      const _Float16* bp = Wfuh + (size_t)(wave*64 + t*16 + l16) * 512 + k0 + hi*16;
      y[t] = WMMA(a, *(const v16h*)bp, y[t]);
    }
  }

  // ---------------- Phase 5: LayerNorm over F=512 + ReLU + store
  float ps[8], pq[8];
#pragma unroll
  for (int r = 0; r < 8; ++r) { ps[r] = 0.0f; pq[r] = 0.0f; }
#pragma unroll
  for (int t = 0; t < 4; ++t)
#pragma unroll
    for (int r = 0; r < 8; ++r) { const float v = y[t][r]; ps[r] += v; pq[r] += v*v; }
#pragma unroll
  for (int r = 0; r < 8; ++r) {
    atomicAdd(&redS[r + rH], ps[r]);   // ds_add_f32
    atomicAdd(&redQ[r + rH], pq[r]);
  }
  __syncthreads();

#pragma unroll
  for (int r = 0; r < 8; ++r) {
    const int row  = r + rH;
    const float mu  = redS[row] * (1.0f/512.0f);
    const float var = redQ[row] * (1.0f/512.0f) - mu*mu;
    const float inv = rsqrtf(var + 1e-5f);
#pragma unroll
    for (int t = 0; t < 4; ++t) {
      const int col = wave*64 + t*16 + l16;
      float v = (y[t][r] - mu) * inv * gc[t] + bc[t];
      v = v > 0.0f ? v : 0.0f;
      out[(size_t)(rowBase + row) * 512 + col] = v;
    }
  }
}

// Prologue: weights f32 -> f16 in workspace (done per launch; deterministic).
__global__ void cvt_f32_f16(const float* __restrict__ s, _Float16* __restrict__ d, int n) {
  int i = blockIdx.x * blockDim.x + threadIdx.x;
  if (i < n) d[i] = (_Float16)s[i];
}

extern "C" void kernel_launch(void* const* d_in, const int* in_sizes, int n_in,
                              void* d_out, int out_size, void* d_ws, size_t ws_size,
                              hipStream_t stream) {
  (void)in_sizes; (void)n_in; (void)out_size; (void)ws_size;
  const float* X   = (const float*)d_in[0];
  const float* W1  = (const float*)d_in[1];
  const float* b1  = (const float*)d_in[2];
  const float* W2  = (const float*)d_in[3];
  const float* b2  = (const float*)d_in[4];
  const float* Wf  = (const float*)d_in[5];
  const float* bf  = (const float*)d_in[6];
  const float* Wfu = (const float*)d_in[7];
  const float* bfu = (const float*)d_in[8];
  const float* lng = (const float*)d_in[9];
  const float* lnb = (const float*)d_in[10];
  // d_in[11] = num_steps (16, hardcoded)

  _Float16* wh = (_Float16*)d_ws;  // needs 720896 halves = 1.44 MB
  cvt_f32_f16<<<(512*512 + 255)/256, 256, 0, stream>>>(W1,  wh,                               512*512);
  cvt_f32_f16<<<(256*512 + 255)/256, 256, 0, stream>>>(W2,  wh + 512*512,                     256*512);
  cvt_f32_f16<<<(256*256 + 255)/256, 256, 0, stream>>>(Wf,  wh + 512*512 + 256*512,           256*256);
  cvt_f32_f16<<<(512*512 + 255)/256, 256, 0, stream>>>(Wfu, wh + 512*512 + 256*512 + 256*256, 512*512);

  // 8192 tokens / 16 rows per workgroup = 512 workgroups, 256 threads (8 waves) each.
  lsn_fused<<<512, 256, 0, stream>>>(X, b1, b2, bf, bfu, lng, lnb, wh, (float*)d_out);
}
